// DustbinSCOTHead_73461120631332
// MI455X (gfx1250) — compile-verified, matrix operand order checked by reading, not run on hardware
//
#include <hip/hip_runtime.h>
#include <hip/hip_bf16.h>
#include <math.h>

// Problem constants (from reference)
#define NROW 32768
#define DDIM 256
#define HDIM 256
#define KP   64
#define NT   17
#define NCOL 1089          // KP*NT + 1 (dustbin)
#define INV_TAU 10.0f
#define SINK_ITERS 50
#define NBLK (NROW / 256)  // 128

typedef __attribute__((ext_vector_type(16))) __bf16 v16bf;
typedef __attribute__((ext_vector_type(8)))  float  v8f;

// ---------------- bf16 helpers (bit-exact, no reliance on cvt patterns) -----
__device__ __forceinline__ unsigned short f2bf_rne(float f) {
  union { float f; unsigned int u; } v; v.f = f;
  unsigned int u = v.u;
  unsigned int r = u + 0x7FFFu + ((u >> 16) & 1u);
  return (unsigned short)(r >> 16);
}
__device__ __forceinline__ float bf2f(unsigned short h) {
  union { unsigned int u; float f; } v; v.u = ((unsigned int)h) << 16; return v.f;
}
__device__ __forceinline__ void split_bf(float x, unsigned short& hi, unsigned short& lo) {
  hi = f2bf_rne(x);
  lo = f2bf_rne(x - bf2f(hi));
}

// ---------------- WMMA fragment load ---------------------------------------
// A-matrix 16x32 bf16 layout (CDNA5 ISA 7.12.2): lane L<16 -> row M=L, K chunks
// {0..7} (VGPR0-3) and {16..23} (VGPR4-7); lanes 16-31 -> K {8..15},{24..31}.
// B is fed K-contiguous from the transposed operand => identical pattern.
__device__ __forceinline__ v16bf load_frag(const unsigned short* p) {
  union { v16bf v; uint4 q[2]; } u;
  u.q[0] = *(const uint4*)(p);        // K k0 .. k0+7
  u.q[1] = *(const uint4*)(p + 16);   // K k0+16 .. k0+23
  return u.v;
}
__device__ __forceinline__ v8f wmma_acc(v16bf a, v16bf b, v8f c) {
  return __builtin_amdgcn_wmma_f32_16x16x32_bf16(false, a, false, b, (short)0, c,
                                                 false, false);
}

// ---------------- prep kernels ---------------------------------------------
__global__ void split_cvt_kernel(const float* __restrict__ src,
                                 unsigned short* __restrict__ hi,
                                 unsigned short* __restrict__ lo, int n) {
  int i = blockIdx.x * blockDim.x + threadIdx.x;
  if (i < n) { unsigned short a, b; split_bf(src[i], a, b); hi[i] = a; lo[i] = b; }
}

__global__ void w1t_kernel(const float* __restrict__ W1,
                           unsigned short* __restrict__ hi,
                           unsigned short* __restrict__ lo) {
  int i = blockIdx.x * blockDim.x + threadIdx.x;   // over HDIM*DDIM = 65536
  int h = i >> 8, d = i & 255;
  unsigned short a, b; split_bf(W1[d * HDIM + h], a, b);
  hi[h * DDIM + d] = a; lo[h * DDIM + d] = b;
}

__global__ void pp_kernel(const float* __restrict__ P, float* __restrict__ pp) {
  int k = threadIdx.x;
  if (k < KP) {
    float s = 0.f;
    for (int j = 0; j < HDIM; ++j) { float x = P[k * HDIM + j]; s += x * x; }
    pp[k] = s;
  }
}

// ---------------- deterministic stable counting sort by joint type ----------
__global__ void hist_kernel(const int* __restrict__ jt, int* __restrict__ blockCnt) {
  __shared__ int c[NT];
  int tid = threadIdx.x;
  if (tid < NT) c[tid] = 0;
  __syncthreads();
  atomicAdd(&c[jt[blockIdx.x * 256 + tid]], 1);
  __syncthreads();
  if (tid < NT) blockCnt[blockIdx.x * NT + tid] = c[tid];
}

__global__ void scan_kernel(const int* __restrict__ blockCnt,
                            int* __restrict__ typeCnt, int* __restrict__ typeOff,
                            int* __restrict__ base) {
  if (threadIdx.x != 0) return;
  int tot[NT];
  for (int t = 0; t < NT; ++t) {
    int s = 0;
    for (int b = 0; b < NBLK; ++b) s += blockCnt[b * NT + t];
    tot[t] = s; typeCnt[t] = s;
  }
  int o = 0;
  for (int t = 0; t < NT; ++t) { typeOff[t] = o; o += tot[t]; }
  for (int t = 0; t < NT; ++t) {
    int r = typeOff[t];
    for (int b = 0; b < NBLK; ++b) { base[b * NT + t] = r; r += blockCnt[b * NT + t]; }
  }
}

__global__ void place_kernel(const int* __restrict__ jt, const int* __restrict__ base,
                             int* __restrict__ rows, int* __restrict__ inv,
                             int* __restrict__ jt_s) {
  __shared__ int sjt[256];
  int tid = threadIdx.x;
  int i = blockIdx.x * 256 + tid;
  int t = jt[i];
  sjt[tid] = t;
  __syncthreads();
  int rank = 0;
  for (int j = 0; j < tid; ++j) rank += (sjt[j] == t) ? 1 : 0;   // stable in-block rank
  int pos = base[blockIdx.x * NT + t] + rank;
  rows[pos] = i; inv[i] = pos; jt_s[pos] = t;
}

// ---------------- GEMM1: h = selu(emb @ W1 + b1), split-bf16 out ------------
// One wave computes a 16(M) x 64(N) tile: A fragments reused across 4 N-tiles.
__global__ void gemm_h_kernel(const unsigned short* __restrict__ Ah,
                              const unsigned short* __restrict__ Al,
                              const unsigned short* __restrict__ Bh,
                              const unsigned short* __restrict__ Bl,
                              const float* __restrict__ b1,
                              unsigned short* __restrict__ Hhi,
                              unsigned short* __restrict__ Hlo) {
  int blk = blockIdx.x;                    // (NROW/16)*(HDIM/64) blocks, 1 wave each
  int ntg = blk & (HDIM / 64 - 1);
  int mt  = blk / (HDIM / 64);
  int lane = threadIdx.x;
  int halfw = lane >> 4, idx = lane & 15;
  const unsigned short* arh = Ah + (size_t)(mt * 16 + idx) * DDIM + halfw * 8;
  const unsigned short* arl = Al + (size_t)(mt * 16 + idx) * DDIM + halfw * 8;
  v8f c[4] = {v8f{}, v8f{}, v8f{}, v8f{}};
#pragma unroll
  for (int ks = 0; ks < DDIM / 32; ++ks) {
    v16bf ah = load_frag(arh + ks * 32);
    v16bf al = load_frag(arl + ks * 32);
#pragma unroll
    for (int q = 0; q < 4; ++q) {
      size_t boff = (size_t)((ntg * 4 + q) * 16 + idx) * DDIM + halfw * 8 + ks * 32;
      v16bf bh = load_frag(Bh + boff);
      v16bf bl = load_frag(Bl + boff);
      c[q] = wmma_acc(ah, bh, c[q]);   // split-bf16: hi*hi + hi*lo + lo*hi ~ fp32
      c[q] = wmma_acc(ah, bl, c[q]);
      c[q] = wmma_acc(al, bh, c[q]);
    }
  }
#pragma unroll
  for (int q = 0; q < 4; ++q) {
    int n = (ntg * 4 + q) * 16 + idx;
    float bias = b1[n];
#pragma unroll
    for (int r = 0; r < 8; ++r) {          // C/D layout: M = r + 8*halfw, N = idx
      int m = mt * 16 + r + halfw * 8;
      float x = c[q][r] + bias;
      float s = (x > 0.0f) ? x : 1.6732632423543772f * (__expf(x) - 1.0f);
      s *= 1.0507009873554805f;
      unsigned short hi, lo; split_bf(s, hi, lo);
      Hhi[(size_t)m * HDIM + n] = hi;
      Hlo[(size_t)m * HDIM + n] = lo;
    }
  }
}

__global__ void hh_kernel(const unsigned short* __restrict__ Hhi,
                          const unsigned short* __restrict__ Hlo,
                          float* __restrict__ hh) {
  int i = blockIdx.x * blockDim.x + threadIdx.x;
  size_t base = (size_t)i * HDIM;
  float s = 0.f;
  for (int j = 0; j < HDIM; ++j) {
    float x = bf2f(Hhi[base + j]) + bf2f(Hlo[base + j]);
    s += x * x;
  }
  hh[i] = s;
}

// ---------------- GEMM2: distT[k][pos] = max(||h||^2+||p||^2-2 h.p, 0) ------
// One wave per 16-row tile covers all KP=64 prototypes (A loaded once).
// Epilogue scatters into the type-sorted transposed layout via inv[].
__global__ void gemm_dist_kernel(const unsigned short* __restrict__ Ah,
                                 const unsigned short* __restrict__ Al,
                                 const unsigned short* __restrict__ Bh,
                                 const unsigned short* __restrict__ Bl,
                                 const float* __restrict__ hh,
                                 const float* __restrict__ pp,
                                 const int* __restrict__ inv,
                                 float* __restrict__ distT) {
  int mt = blockIdx.x;                     // NROW/16 blocks, 1 wave each
  int lane = threadIdx.x;
  int halfw = lane >> 4, idx = lane & 15;
  const unsigned short* arh = Ah + (size_t)(mt * 16 + idx) * HDIM + halfw * 8;
  const unsigned short* arl = Al + (size_t)(mt * 16 + idx) * HDIM + halfw * 8;
  v8f c[4] = {v8f{}, v8f{}, v8f{}, v8f{}};
#pragma unroll
  for (int ks = 0; ks < HDIM / 32; ++ks) {
    v16bf ah = load_frag(arh + ks * 32);
    v16bf al = load_frag(arl + ks * 32);
#pragma unroll
    for (int q = 0; q < 4; ++q) {
      size_t boff = (size_t)(q * 16 + idx) * HDIM + halfw * 8 + ks * 32;
      v16bf bh = load_frag(Bh + boff);
      v16bf bl = load_frag(Bl + boff);
      c[q] = wmma_acc(ah, bh, c[q]);
      c[q] = wmma_acc(ah, bl, c[q]);
      c[q] = wmma_acc(al, bh, c[q]);
    }
  }
#pragma unroll
  for (int r = 0; r < 8; ++r) {
    int m = mt * 16 + r + halfw * 8;
    float hhm = hh[m];
    int pos = inv[m];
#pragma unroll
    for (int q = 0; q < 4; ++q) {
      int n = q * 16 + idx;                // n == prototype index k
      float d = hhm + pp[n] - 2.0f * c[q][r];
      distT[(size_t)n * NROW + pos] = fmaxf(d, 0.0f);
    }
  }
}

// ---------------- Sinkhorn ---------------------------------------------------
__global__ void init_uv_kernel(float* log_u_s, float* log_v) {
  int i = blockIdx.x * blockDim.x + threadIdx.x;
  if (i < NROW) log_u_s[i] = 0.0f;
  if (i < NCOL) log_v[i] = 0.0f;
}

// Row pass over sorted positions: fully coalesced distT[k*N+pos] reads.
// Only the 64 slots at t=jt[i] plus dustbin survive in fp32 (masked = exact 0).
__global__ void sink_row_kernel(const float* __restrict__ distT,
                                const int* __restrict__ jt_s,
                                const float* __restrict__ log_v,
                                const float* __restrict__ dc,
                                float* __restrict__ log_u_s) {
  __shared__ float lv[NCOL];
  for (int j = threadIdx.x; j < NCOL; j += blockDim.x) lv[j] = log_v[j];
  __syncthreads();
  int pos = blockIdx.x * blockDim.x + threadIdx.x;
  float lkd = -fabsf(dc[0]) * INV_TAU;
  int t = jt_s[pos];
  float dterm = lkd + lv[KP * NT];
  float m = dterm;
  for (int k = 0; k < KP; ++k) {
    float x = -distT[(size_t)k * NROW + pos] * INV_TAU + lv[k * NT + t];
    m = fmaxf(m, x);
  }
  float s = __expf(dterm - m);
  for (int k = 0; k < KP; ++k) {
    float x = -distT[(size_t)k * NROW + pos] * INV_TAU + lv[k * NT + t];
    s += __expf(x - m);
  }
  log_u_s[pos] = -(m + __logf(s));
}

__device__ __forceinline__ float block_max(float v, float* red) {
  int tid = threadIdx.x;
  red[tid] = v; __syncthreads();
  for (int st = 128; st > 0; st >>= 1) {
    if (tid < st) red[tid] = fmaxf(red[tid], red[tid + st]);
    __syncthreads();
  }
  float r = red[0]; __syncthreads();
  return r;
}
__device__ __forceinline__ float block_sum(float v, float* red) {
  int tid = threadIdx.x;
  red[tid] = v; __syncthreads();
  for (int st = 128; st > 0; st >>= 1) {
    if (tid < st) red[tid] += red[tid + st];
    __syncthreads();
  }
  float r = red[0]; __syncthreads();
  return r;
}

// Column pass: one block per column j; contiguous slice of distT / log_u_s.
__global__ void sink_col_kernel(const float* __restrict__ distT,
                                const float* __restrict__ log_u_s,
                                const int* __restrict__ typeCnt,
                                const int* __restrict__ typeOff,
                                const float* __restrict__ dc,
                                float* __restrict__ log_v) {
  __shared__ float red[256];
  int j = blockIdx.x;
  if (j == KP * NT) {                 // dustbin: logsumexp over all rows of log_u
    float m = -INFINITY;
    for (int r = threadIdx.x; r < NROW; r += blockDim.x) m = fmaxf(m, log_u_s[r]);
    m = block_max(m, red);
    float s = 0.f;
    for (int r = threadIdx.x; r < NROW; r += blockDim.x) s += __expf(log_u_s[r] - m);
    s = block_sum(s, red);
    if (threadIdx.x == 0) {
      float lkd = -fabsf(dc[0]) * INV_TAU;
      float log_s = lkd + m + __logf(s);
      log_v[j] = __logf((float)NROW) - log_s;
    }
  } else {                            // column (k,t): only rows with jt==t survive
    int k = j / NT, t = j % NT;
    int o = typeOff[t], c = typeCnt[t];
    const float* dk = distT + (size_t)k * NROW + o;
    const float* us = log_u_s + o;
    float m = -INFINITY;
    for (int q = threadIdx.x; q < c; q += blockDim.x)
      m = fmaxf(m, -dk[q] * INV_TAU + us[q]);
    m = block_max(m, red);
    float s = 0.f;
    for (int q = threadIdx.x; q < c; q += blockDim.x)
      s += __expf(-dk[q] * INV_TAU + us[q] - m);
    s = block_sum(s, red);
    if (threadIdx.x == 0) log_v[j] = -(m + __logf(s));
  }
}

// ---------------- output -----------------------------------------------------
__global__ void zero4_kernel(float4* __restrict__ p, int n4) {
  int i = blockIdx.x * blockDim.x + threadIdx.x;
  if (i < n4) p[i] = make_float4(0.f, 0.f, 0.f, 0.f);
}

__global__ void final_kernel(const float* __restrict__ distT,
                             const int* __restrict__ rows,
                             const int* __restrict__ jt_s,
                             const float* __restrict__ log_u_s,
                             const float* __restrict__ log_v,
                             const float* __restrict__ dc,
                             float* __restrict__ logits,
                             float* __restrict__ T) {
  __shared__ float lv[NCOL];
  for (int j = threadIdx.x; j < NCOL; j += blockDim.x) lv[j] = log_v[j];
  __syncthreads();
  int pos = blockIdx.x * blockDim.x + threadIdx.x;
  int i = rows[pos];
  float u = log_u_s[pos];
  int t = jt_s[pos];
  float lkd = -fabsf(dc[0]) * INV_TAU;
  size_t tb = (size_t)i * NCOL;
  T[tb + KP * NT] = __expf(lkd + u + lv[KP * NT]);
  for (int k = 0; k < KP; ++k) {
    int j = k * NT + t;
    float Tv = __expf(-distT[(size_t)k * NROW + pos] * INV_TAU + u + lv[j]);
    T[tb + j] = Tv;
    // person_mass[i,k] = Tv (the other 16 type-slots are exact fp32 zeros)
    logits[(size_t)i * KP + k] = __logf(Tv + 1e-8f);
  }
}

// ---------------- host launcher ---------------------------------------------
extern "C" void kernel_launch(void* const* d_in, const int* in_sizes, int n_in,
                              void* d_out, int out_size, void* d_ws, size_t ws_size,
                              hipStream_t stream) {
  const float* emb = (const float*)d_in[0];
  const float* W1  = (const float*)d_in[1];
  const float* b1  = (const float*)d_in[2];
  const float* P   = (const float*)d_in[3];
  const float* dc  = (const float*)d_in[4];
  const int*   jt  = (const int*)d_in[5];
  // d_in[6] = k (== KP, compile-time here)

  char* ws = (char*)d_ws;
  size_t off = 0;
  auto take = [&](size_t bytes) -> void* {
    void* p = (void*)(ws + off);
    off += (bytes + 255) & ~(size_t)255;
    return p;
  };

  unsigned short* embHi = (unsigned short*)take((size_t)NROW * DDIM * 2);
  unsigned short* embLo = (unsigned short*)take((size_t)NROW * DDIM * 2);
  unsigned short* w1tHi = (unsigned short*)take((size_t)HDIM * DDIM * 2);
  unsigned short* w1tLo = (unsigned short*)take((size_t)HDIM * DDIM * 2);
  unsigned short* pHi   = (unsigned short*)take((size_t)KP * HDIM * 2);
  unsigned short* pLo   = (unsigned short*)take((size_t)KP * HDIM * 2);
  float*          ppv   = (float*)take(KP * 4);
  unsigned short* hHi   = (unsigned short*)take((size_t)NROW * HDIM * 2);
  unsigned short* hLo   = (unsigned short*)take((size_t)NROW * HDIM * 2);
  float*          hh    = (float*)take((size_t)NROW * 4);
  float*          distT = (float*)take((size_t)NROW * KP * 4);
  float*          logU  = (float*)take((size_t)NROW * 4);
  float*          logV  = (float*)take((size_t)NCOL * 4);
  int*            blkC  = (int*)take((size_t)NBLK * NT * 4);
  int*            baseA = (int*)take((size_t)NBLK * NT * 4);
  int*            tCnt  = (int*)take(NT * 4);
  int*            tOff  = (int*)take(NT * 4);
  int*            rows  = (int*)take((size_t)NROW * 4);
  int*            inv   = (int*)take((size_t)NROW * 4);
  int*            jt_s  = (int*)take((size_t)NROW * 4);

  float* logits = (float*)d_out;
  float* T      = logits + (size_t)NROW * KP;

  // --- prep: split-bf16 conversions, transpose, norms ---
  split_cvt_kernel<<<(NROW * DDIM + 255) / 256, 256, 0, stream>>>(emb, embHi, embLo, NROW * DDIM);
  split_cvt_kernel<<<(KP * HDIM + 255) / 256, 256, 0, stream>>>(P, pHi, pLo, KP * HDIM);
  w1t_kernel<<<(HDIM * DDIM + 255) / 256, 256, 0, stream>>>(W1, w1tHi, w1tLo);
  pp_kernel<<<1, 64, 0, stream>>>(P, ppv);

  // --- deterministic stable counting sort of rows by joint type ---
  hist_kernel<<<NBLK, 256, 0, stream>>>(jt, blkC);
  scan_kernel<<<1, 1, 0, stream>>>(blkC, tCnt, tOff, baseA);
  place_kernel<<<NBLK, 256, 0, stream>>>(jt, baseA, rows, inv, jt_s);

  // --- WMMA GEMMs (split-bf16, 16x64 tiles per wave) ---
  gemm_h_kernel<<<(NROW / 16) * (HDIM / 64), 32, 0, stream>>>(embHi, embLo, w1tHi, w1tLo,
                                                              b1, hHi, hLo);
  hh_kernel<<<NROW / 256, 256, 0, stream>>>(hHi, hLo, hh);
  gemm_dist_kernel<<<NROW / 16, 32, 0, stream>>>(hHi, hLo, pHi, pLo, hh, ppv, inv, distT);

  // --- Sinkhorn (active-structure exact in fp32; distT stays L2-resident,
  //     both passes fully coalesced in the type-sorted transposed layout) ---
  init_uv_kernel<<<NROW / 256, 256, 0, stream>>>(logU, logV);
  for (int it = 0; it < SINK_ITERS; ++it) {
    sink_row_kernel<<<NROW / 256, 256, 0, stream>>>(distT, jt_s, logV, dc, logU);
    sink_col_kernel<<<NCOL, 256, 0, stream>>>(distT, logU, tCnt, tOff, dc, logV);
  }

  // --- outputs: zero T, scatter active entries, logits ---
  int n4 = (NROW * NCOL) / 4;   // 1089*32768 divisible by 4
  zero4_kernel<<<(n4 + 255) / 256, 256, 0, stream>>>((float4*)T, n4);
  final_kernel<<<NROW / 256, 256, 0, stream>>>(distT, rows, jt_s, logU, logV, dc, logits, T);
}
